// FrameSoftAttention_22402549416200
// MI455X (gfx1250) — compile-verified, matrix operand order checked by reading, not run on hardware
//
#include <hip/hip_runtime.h>

// FrameSoftAttention for MI455X (gfx1250, wave32, WMMA + async/TDM staging).
// B=128, I=J=1024, D_IN=D_INNER=256.
//
// Pass 1 (x3): qh/kh/vh = X @ W^T, f32 in -> f16 out (workspace, 3*64MB).
//              W staged through LDS as f16 k-slabs (converted once per block).
// Pass 2: flash attention per (i-block, batch):
//         K block staged via global_load_async_to_lds_b128 (ASYNCcnt),
//         V block staged via TDM tensor_load_to_lds (TENSORcnt) + LDS transpose,
//         S = Q K^T (WMMA f16), online softmax (f32), O += P V (WMMA f16).
// All WMMA inner loops software-pipeline their LDS B-fragment loads so ds
// latency overlaps matrix issue instead of draining DScnt to 0 per fragment.

typedef __attribute__((ext_vector_type(16))) _Float16     v16h;
typedef __attribute__((ext_vector_type(8)))  float        v8f;
typedef __attribute__((ext_vector_type(4)))  unsigned int su4;
typedef __attribute__((ext_vector_type(8)))  unsigned int su8;

union HF16 { v16h v; _Float16 h[16]; uint4 q[2]; };

__device__ __forceinline__ v8f wmma16x16x32(v16h a, v16h b, v8f c) {
  // D = A(16x32 f16) * B(32x16 f16) + C(16x16 f32)
  return __builtin_amdgcn_wmma_f32_16x16x32_f16(false, a, false, b,
                                                (short)0, c, false, false);
}

// 16-bit A/B fragment of a 16x32 tile from a row-major f16 matrix.
// Per-lane K pattern (ISA 7.12.2): halves {g*8..g*8+7, 16+g*8..16+g*8+7}.
__device__ __forceinline__ v16h frag_f16(const _Float16* base, int ld, int row,
                                         int kb, int g) {
  HF16 f;
  const _Float16* p = base + (size_t)row * ld + kb + (g << 3);
  f.q[0] = *(const uint4*)(p);
  f.q[1] = *(const uint4*)(p + 16);
  return f.v;
}

// Same fragment but sourced from f32 memory (converted on the fly).
__device__ __forceinline__ v16h frag_f32(const float* base, int ld, int row,
                                         int kb, int g) {
  HF16 f;
  const float* p = base + (size_t)row * ld + kb + (g << 3);
#pragma unroll
  for (int i = 0; i < 8; ++i) f.h[i] = (_Float16)p[i];
#pragma unroll
  for (int i = 0; i < 8; ++i) f.h[8 + i] = (_Float16)p[16 + i];
  return f.v;
}

// ---------------------------------------------------------------------------
// Pass 1: Y[M,256] (f16) = X[M,256] (f32) @ W[256,256]^T (f32)
// Block = 256 threads = 8 waves; each wave owns a 16-row strip.
// W k-slab (256 rows x 32 k) converted to f16 in LDS once per k-step.
// ---------------------------------------------------------------------------
__global__ __launch_bounds__(256) void proj_xwt_f16(
    const float* __restrict__ X, const float* __restrict__ W,
    _Float16* __restrict__ Y) {
  __shared__ __align__(16) _Float16 lW[256 * 32];  // [n][k] slab, 16KB

  const int t    = threadIdx.x;
  const int lane = t & 31;
  const int wave = t >> 5;
  const int l16  = lane & 15;
  const int g    = lane >> 4;
  const int rowBase = blockIdx.x * 128 + wave * 16;

  // Preload all 8 A fragments (one 16-row strip, K = 0..255).
  v16h a[8];
#pragma unroll
  for (int s = 0; s < 8; ++s)
    a[s] = frag_f32(X, 256, rowBase + l16, s * 32, g);

  // All 16 output n-tiles live across the k loop (128 acc VGPRs).
  v8f acc[16];
  {
    v8f z = {};
#pragma unroll
    for (int nt = 0; nt < 16; ++nt) acc[nt] = z;
  }

  for (int s = 0; s < 8; ++s) {
    __syncthreads();  // protect previous slab from overwrite
    {
      // Thread t converts row n=t of the k-slab: 32 f32 -> 32 f16.
      const float* wr = W + (size_t)t * 256 + s * 32;
      _Float16* dst = lW + t * 32;
#pragma unroll
      for (int i = 0; i < 32; ++i) dst[i] = (_Float16)wr[i];
    }
    __syncthreads();
    // Software-pipelined: load fragment nt+1 while WMMA consumes nt.
    v16h b = frag_f16(lW, 32, l16, 0, g);
#pragma unroll
    for (int nt = 0; nt < 15; ++nt) {
      v16h bn = frag_f16(lW, 32, (nt + 1) * 16 + l16, 0, g);
      acc[nt] = wmma16x16x32(a[s], b, acc[nt]);
      b = bn;
    }
    acc[15] = wmma16x16x32(a[s], b, acc[15]);
  }

  // C layout: VGPR r -> row (rowBase + r + 8*g), col nt*16 + l16.
#pragma unroll
  for (int nt = 0; nt < 16; ++nt) {
    const int n = nt * 16 + l16;
#pragma unroll
    for (int r = 0; r < 8; ++r)
      Y[(size_t)(rowBase + r + 8 * g) * 256 + n] = (_Float16)acc[nt][r];
  }
}

// ---------------------------------------------------------------------------
// Pass 2: flash attention.
// Grid: (8 i-blocks, 128 batches). Block = 256 threads = 8 waves.
// ---------------------------------------------------------------------------
__global__ __launch_bounds__(256) void flash_attn(
    const _Float16* __restrict__ qh, const _Float16* __restrict__ kh,
    const _Float16* __restrict__ vh, float* __restrict__ out) {
  __shared__ __align__(16) _Float16 lK[32 * 256];    // K row-major (16KB)
  __shared__ __align__(16) _Float16 lVrow[32 * 256]; // V row-major (16KB, TDM)
  __shared__ __align__(16) _Float16 lVt[256 * 32];   // V transposed (16KB)
  __shared__ __align__(16) _Float16 lP[8 * 16 * 32]; // per-wave P staging (8KB)

  const int t    = threadIdx.x;
  const int lane = t & 31;
  const int wave = t >> 5;
  const int l16  = lane & 15;
  const int g    = lane >> 4;
  const int b    = blockIdx.y;
  const int iBase = blockIdx.x * 128 + wave * 16;

  const _Float16* qp  = qh + ((size_t)b * 1024 + iBase) * 256;
  const _Float16* kbp = kh + (size_t)b * 1024 * 256;
  const _Float16* vbp = vh + (size_t)b * 1024 * 256;
  _Float16* lPw = lP + wave * 512;

  const unsigned int lKbase = (unsigned int)(unsigned long long)(const void*)lK;
  const unsigned int lVbase = (unsigned int)(unsigned long long)(const void*)lVrow;

  // Q strip as 8 resident A fragments (64 VGPRs).
  v16h qf[8];
#pragma unroll
  for (int s = 0; s < 8; ++s)
    qf[s] = frag_f16(qp, 256, l16, s * 32, g);

  // O accumulator: 16 n-tiles x (16x16 f32 C tile) = 128 VGPRs.
  v8f o[16];
  {
    v8f z = {};
#pragma unroll
    for (int nt = 0; nt < 16; ++nt) o[nt] = z;
  }
  float mrow[8], lrow[8];
#pragma unroll
  for (int r = 0; r < 8; ++r) { mrow[r] = -1.0e30f; lrow[r] = 0.0f; }

  const float scale = 0.0625f;  // 256^-0.5

  for (int jb = 0; jb < 1024; jb += 32) {
    // --- stage K block: per-lane async copy global -> LDS (ASYNCcnt) ---
    {
      const unsigned long long gK =
          (unsigned long long)(const void*)(kbp + (size_t)jb * 256);
#pragma unroll
      for (int it = 0; it < 4; ++it) {
        const int idx = t + it * 256;  // 16-byte chunk index, 1024 total
        unsigned long long ga = gK + (unsigned long long)idx * 16u;
        unsigned int la = lKbase + (unsigned int)idx * 16u;
        asm volatile("global_load_async_to_lds_b128 %0, %1, off"
                     :: "v"(la), "v"(ga) : "memory");
      }
    }
    // --- stage V block: TDM tensor load, one descriptor from wave 0 ---
    if (t < 32) {
      const unsigned long long gV =
          (unsigned long long)(const void*)(vbp + (size_t)jb * 256);
      su4 g0;
      su8 g1;
      g0[0] = 1u;                                  // count=1 user descriptor
      g0[1] = lVbase;                              // lds_addr
      g0[2] = (unsigned int)gV;                    // global_addr[31:0]
      g0[3] = (unsigned int)((gV >> 32) & 0x01FFFFFFu) | (2u << 30); // +type=2
      g1[0] = (1u << 16);        // workgroup_mask=0, data_size=1 (2 bytes)
      g1[1] = (8192u << 16);     // tensor_dim0[15:0] = 8192 elements
      g1[2] = (1u << 16);        // tensor_dim1 = 1
      g1[3] = (8192u << 16);     // tile_dim0 = 8192 (one contiguous tile)
      g1[4] = 1u;                // tile_dim1 = 1, tile_dim2 = 0
      g1[5] = 8192u;             // tensor_dim0_stride[31:0]
      g1[6] = 0u;
      g1[7] = 0u;
      asm volatile("tensor_load_to_lds %0, %1" :: "s"(g0), "s"(g1) : "memory");
      __builtin_amdgcn_s_wait_tensorcnt(0);
    }
    asm volatile("s_wait_asynccnt 0x0" ::: "memory");
    if (jb + 32 < 1024)  // keep next K tile warm in L2
      __builtin_prefetch(kbp + (size_t)(jb + 32) * 256 + lane * 64, 0, 0);
    __syncthreads();

    // --- LDS->LDS transpose of V: lVrow[32][256] -> lVt[256][32] ---
    {
      const int jl = t & 31;   // local j row
      const int cc = t >> 5;   // 32-column chunk
      const _Float16* vr = lVrow + (size_t)jl * 256 + cc * 32;
#pragma unroll
      for (int i = 0; i < 32; ++i)
        lVt[(cc * 32 + i) * 32 + jl] = vr[i];
    }
    __syncthreads();

    // --- S = Q K^T : two 16x16 C tiles, pipelined K-fragment loads ---
    v8f s0 = {}, s1 = {};
    v16h b0 = frag_f16(lK, 256, l16, 0, g);
    v16h b1 = frag_f16(lK, 256, 16 + l16, 0, g);
#pragma unroll
    for (int s = 0; s < 7; ++s) {
      v16h n0 = frag_f16(lK, 256, l16, (s + 1) * 32, g);
      v16h n1 = frag_f16(lK, 256, 16 + l16, (s + 1) * 32, g);
      s0 = wmma16x16x32(qf[s], b0, s0);
      s1 = wmma16x16x32(qf[s], b1, s1);
      b0 = n0;
      b1 = n1;
    }
    s0 = wmma16x16x32(qf[7], b0, s0);
    s1 = wmma16x16x32(qf[7], b1, s1);

    // --- online softmax over the 32 new columns (per C-layout row) ---
#pragma unroll
    for (int r = 0; r < 8; ++r) {
      float a0 = s0[r] * scale;
      float a1 = s1[r] * scale;
      float mx = fmaxf(a0, a1);
#pragma unroll
      for (int off = 1; off < 16; off <<= 1)
        mx = fmaxf(mx, __shfl_xor(mx, off, 32));
      const float mnew = fmaxf(mrow[r], mx);
      const float al   = __expf(mrow[r] - mnew);
      const float p0   = __expf(a0 - mnew);
      const float p1   = __expf(a1 - mnew);
      float rs = p0 + p1;
#pragma unroll
      for (int off = 1; off < 16; off <<= 1)
        rs += __shfl_xor(rs, off, 32);
      lrow[r] = lrow[r] * al + rs;
      mrow[r] = mnew;
#pragma unroll
      for (int nt = 0; nt < 16; ++nt) o[nt][r] *= al;
      lPw[(r + 8 * g) * 32 + l16]      = (_Float16)p0;
      lPw[(r + 8 * g) * 32 + 16 + l16] = (_Float16)p1;
    }

    // P as an A fragment (wave-private LDS round trip, C->A layout swap).
    v16h pf = frag_f16(lPw, 32, l16, 0, g);

    // --- O += P V : 16 n-tiles, pipelined V-fragment loads ---
    v16h vf = frag_f16(lVt, 32, l16, 0, g);
#pragma unroll
    for (int nt = 0; nt < 15; ++nt) {
      v16h vn = frag_f16(lVt, 32, (nt + 1) * 16 + l16, 0, g);
      o[nt] = wmma16x16x32(pf, vf, o[nt]);
      vf = vn;
    }
    o[15] = wmma16x16x32(pf, vf, o[15]);
    __syncthreads();  // protect lK/lVrow/lVt before next stage
  }

  // --- epilogue: normalize and store f32 output ---
  float* op = out + ((size_t)b * 1024 + iBase) * 256;
#pragma unroll
  for (int r = 0; r < 8; ++r) {
    const float inv = 1.0f / lrow[r];
#pragma unroll
    for (int nt = 0; nt < 16; ++nt)
      op[(size_t)(r + 8 * g) * 256 + nt * 16 + l16] = o[nt][r] * inv;
  }
}

// ---------------------------------------------------------------------------
extern "C" void kernel_launch(void* const* d_in, const int* in_sizes, int n_in,
                              void* d_out, int out_size, void* d_ws,
                              size_t ws_size, hipStream_t stream) {
  (void)in_sizes; (void)n_in; (void)out_size; (void)ws_size;
  const float* q  = (const float*)d_in[0];
  const float* k  = (const float*)d_in[1];
  const float* v  = (const float*)d_in[2];
  const float* Wq = (const float*)d_in[3];
  const float* Wk = (const float*)d_in[4];
  const float* Wv = (const float*)d_in[5];
  float* out = (float*)d_out;

  // Workspace: qh/kh/vh in f16, 128*1024*256 halves each (64MB x3 = 192MB).
  const size_t n1 = (size_t)128 * 1024 * 256;
  _Float16* qh  = (_Float16*)d_ws;
  _Float16* khp = qh + n1;
  _Float16* vhp = khp + n1;

  dim3 pgrid(1024);  // 131072 rows / 128 per block
  proj_xwt_f16<<<pgrid, 256, 0, stream>>>(q, Wq, qh);
  proj_xwt_f16<<<pgrid, 256, 0, stream>>>(k, Wk, khp);
  proj_xwt_f16<<<pgrid, 256, 0, stream>>>(v, Wv, vhp);

  dim3 agrid(8, 128);  // (i-blocks, batch)
  flash_attn<<<agrid, 256, 0, stream>>>(qh, khp, vhp, out);
}
